// LSTMLayer_26637387170537
// MI455X (gfx1250) — matrix operand order repaired and checked by reading, not compiled, not run on hardware
//
#include <hip/hip_runtime.h>
#include <hip/hip_bf16.h>

// LSTM layer, T=128, B=128, I=256, H=256, distinct weights per timestep.
//
// Phase 0: init f16 h state (double-buffered), f32 c state, grid-sync counter.
// Phase 1a: pack W_hidden (f32 row-major) -> Whp (f16, WMMA-B register order).
// Phase 1b: pre-gates Gp[t] = x_t @ W_input[t] + b_i[t] + b_h[t], stored in
//           lane-major tile order so phase 2 reloads them as contiguous v8f.
// Phase 2:  persistent kernel, 128 sequential steps with an atomic grid
//           barrier; per step each of 128 waves computes one 16x16 (batch x
//           hidden) tile across all 4 gate blocks (32 v_wmma/wave/step) with
//           only coalesced b128 loads in the serial loop, prefetches the next
//           step's weights during compute, then fuses the LSTM elementwise.

#define T_  128
#define B_  128
#define I_  256
#define H_  256
#define G4H 1024  // 4*H

typedef __attribute__((ext_vector_type(16))) _Float16 v16h;
typedef __attribute__((ext_vector_type(8)))  _Float16 v8h;
typedef __attribute__((ext_vector_type(8)))  float    v8f;

// workspace layout (bytes); total ~134.5 MB
#define WS_GP_BYTES  ((size_t)T_ * B_ * G4H * 4)            // 67,108,864
#define WS_WHP_BYTES ((size_t)T_ * H_ * G4H * 2)            // 67,108,864
#define WS_H16_BYTES ((size_t)2 * B_ * H_ * 2)              // 131,072
#define WS_C_BYTES   ((size_t)B_ * H_ * 4)                  // 131,072
#define WS_WHP_OFF   (WS_GP_BYTES)
#define WS_H16_OFF   (WS_GP_BYTES + WS_WHP_BYTES)
#define WS_C_OFF     (WS_H16_OFF + WS_H16_BYTES)
#define WS_CNT_OFF   (WS_C_OFF + WS_C_BYTES)

__device__ __forceinline__ float sigmoid_(float x) {
    return 1.0f / (1.0f + __expf(-x));
}
// tanh via sigmoid identity: one v_exp + rcp + fma; saturates cleanly at +-1
__device__ __forceinline__ float tanh_(float x) {
    return 2.0f / (1.0f + __expf(-2.0f * x)) - 1.0f;
}

// ---------------------------------------------------------------- phase 0
__global__ void lstm_init_kernel(const float* __restrict__ h0,
                                 const float* __restrict__ c0,
                                 _Float16* __restrict__ h16,
                                 float* __restrict__ cbuf,
                                 unsigned* __restrict__ cnt) {
    int idx = blockIdx.x * blockDim.x + threadIdx.x;
    if (idx < B_ * H_) {
        h16[idx] = (_Float16)h0[idx];   // buffer 0 (t=0 reads this)
        cbuf[idx] = c0[idx];
    }
    if (idx == 0) *cnt = 0u;
}

// ---------------------------------------------------------------- phase 1a
// Pack W_hidden into f16 WMMA-B register order.
// Tile id = ((t*16 + hn)*4 + q)*8 + kc ; one wave per tile; lane L holds
// column n = L%16, K = kc*32 + (L/16)*16 + i (i = 0..15) -> v16h at
// Whp[tile*32 + L]. 65536 tiles, 4 waves/block.
__global__ __launch_bounds__(128) void lstm_pack_wh_kernel(
    const float* __restrict__ Wh,    // [T,H,4H] f32 row-major
    v16h* __restrict__ Whp)          // [65536][32] v16h
{
    const int lane = threadIdx.x & 31;
    const int wave = threadIdx.x >> 5;
    const int tile = blockIdx.x * 4 + wave;   // 0..65535
    const int t  = tile >> 9;
    const int r  = tile & 511;
    const int hn = r >> 5;                    // 0..15
    const int q  = (r >> 3) & 3;              // 0..3
    const int kc = r & 7;                     // 0..7

    const float* src = Wh + (size_t)t * H_ * G4H
                          + (size_t)(kc * 32 + (lane >> 4) * 16) * G4H
                          + q * H_ + hn * 16 + (lane & 15);
    v16h b;
#pragma unroll
    for (int i = 0; i < 16; ++i) b[i] = (_Float16)src[(size_t)i * G4H];
    Whp[(size_t)tile * 32 + lane] = b;
}

// ---------------------------------------------------------------- phase 1b
// One wave per 16x16 tile of the pre-gate matrix (T*8*64 = 65536 tiles).
// Output stored lane-major: Gp[((t*8+bm)*64 + jn)*256 + lane*8 + v].
__global__ __launch_bounds__(128) void lstm_pregate_kernel(
    const float* __restrict__ x,    // [T,B,I]
    const float* __restrict__ Wi,   // [T,I,4H]
    const float* __restrict__ bi,   // [T,4H]
    const float* __restrict__ bh,   // [T,4H]
    float* __restrict__ Gp)         // lane-major tiles
{
    const int lane = threadIdx.x & 31;
    const int wave = threadIdx.x >> 5;
    const int tile = blockIdx.x * 4 + wave;   // 0..65535
    const int t  = tile >> 9;
    const int r  = tile & 511;
    const int bm = r >> 6;                    // batch tile   0..7
    const int jn = r & 63;                    // column tile  0..63

    const int m  = lane & 15;
    const int hi = lane >> 4;
    const int n  = lane & 15;

    const float* xrow  = x + ((size_t)t * B_ + bm * 16 + m) * I_;
    const float* wbase = Wi + (size_t)t * I_ * G4H + jn * 16 + n;
    const int    j     = jn * 16 + n;
    const float  bias  = bi[t * G4H + j] + bh[t * G4H + j];

    v8f acc;
#pragma unroll
    for (int v = 0; v < 8; ++v) acc[v] = bias;  // bias folded into C

    for (int k0 = 0; k0 < I_; k0 += 32) {
        // A 16x32 f16: lanes 0-15 K=0..15, lanes 16-31 K=16..31
        v16h a;
#pragma unroll
        for (int i = 0; i < 8; ++i) a[i]     = (_Float16)xrow[k0 + hi * 16 + i];
#pragma unroll
        for (int i = 0; i < 8; ++i) a[8 + i] = (_Float16)xrow[k0 + hi * 16 + 8 + i];
        // B 32x16 f16: lane = column n, lanes 0-15 K=0..15, lanes 16-31 K=16..31
        v16h b;
#pragma unroll
        for (int i = 0; i < 16; ++i)
            b[i] = (_Float16)wbase[(size_t)(k0 + hi * 16 + i) * G4H];
        acc = __builtin_amdgcn_wmma_f32_16x16x32_f16(false, a, false, b,
                                                     (short)0, acc, false, false);
    }

    float* gout = Gp + (((size_t)t * 8 + bm) * 64 + jn) * 256 + lane * 8;
    *(v8f*)gout = acc;   // 2x global_store_b128, fully coalesced
}

// ---------------------------------------------------------------- phase 2
// Persistent: 32 blocks x 128 threads = 128 waves; wave owns (bm, hn).
__global__ __launch_bounds__(128) void lstm_steps_kernel(
    const float* __restrict__ Gp,    // lane-major pre-gate tiles (biases in)
    const v16h* __restrict__ Whp,    // packed f16 B tiles
    _Float16* __restrict__ h16,      // [2][B*H] f16 state, double-buffered
    float* __restrict__ cbuf,        // [B*H]
    float* __restrict__ out,         // outputs[T*B*H] ++ h[B*H] ++ c[B*H]
    unsigned* __restrict__ cnt)
{
    const int lane = threadIdx.x & 31;
    const int wave = threadIdx.x >> 5;
    const int wg   = blockIdx.x * 4 + wave;   // 0..127
    const int bm   = wg >> 4;                 // batch tile  0..7
    const int hn   = wg & 15;                 // hidden tile 0..15
    const int m    = lane & 15;
    const int hi   = lane >> 4;
    const int n    = lane & 15;

    for (int t = 0; t < T_; ++t) {
        const _Float16* hin  = h16 + (size_t)(t & 1) * (B_ * H_);
        _Float16*       hout = h16 + (size_t)((t + 1) & 1) * (B_ * H_);

        // C init = pre-gates; contiguous v8f per lane per gate block
        v8f acc[4];
        const float* gbase = Gp + (((size_t)t * 8 + bm) * 64 + hn) * 256 + lane * 8;
#pragma unroll
        for (int q = 0; q < 4; ++q)
            acc[q] = *(const v8f*)(gbase + (size_t)q * 16 * 256);

        // Prefetch next step's weights & pre-gates (no dependency on the
        // barrier; pure latency hiding -> global_prefetch_b8).
        {
            const int tn = (t + 1 < T_) ? (t + 1) : t;
            const v16h* wnext = Whp + (((size_t)tn * 16 + hn) * 4) * 8 * 32 + lane;
#pragma unroll
            for (int qq = 0; qq < 4; ++qq)
#pragma unroll
                for (int kk = 0; kk < 8; ++kk)
                    __builtin_prefetch((const void*)(wnext + (size_t)(qq * 8 + kk) * 32), 0, 1);
            const float* gnext = Gp + (((size_t)tn * 8 + bm) * 64 + hn) * 256 + lane * 8;
#pragma unroll
            for (int qq = 0; qq < 4; ++qq)
                __builtin_prefetch((const void*)(gnext + (size_t)qq * 16 * 256), 0, 1);
        }

        const _Float16* hrow  = hin + (size_t)(bm * 16 + m) * H_;
        // B tiles for this wave: tiles ((t*16 + hn)*4 + q)*8 + kc
        const v16h* wbase = Whp + (((size_t)t * 16 + hn) * 4) * 8 * 32 + lane;

#pragma unroll
        for (int kc = 0; kc < 8; ++kc) {
            const int k0 = kc * 32;
            v8h a0 = *(const v8h*)(hrow + k0 + hi * 16);       // b128
            v8h a1 = *(const v8h*)(hrow + k0 + hi * 16 + 8);   // b128
            v16h a;
#pragma unroll
            for (int i = 0; i < 8; ++i) { a[i] = a0[i]; a[8 + i] = a1[i]; }
#pragma unroll
            for (int q = 0; q < 4; ++q) {
                v16h b = wbase[(size_t)(q * 8 + kc) * 32];     // 2x b128, coalesced
                acc[q] = __builtin_amdgcn_wmma_f32_16x16x32_f16(
                    false, a, false, b, (short)0, acc[q], false, false);
            }
        }

        // fused LSTM elementwise; gate order per reference: i, f, o, g
#pragma unroll
        for (int v = 0; v < 8; ++v) {
            const int    brow = bm * 16 + v + 8 * hi;
            const int    col  = hn * 16 + n;
            const size_t idx  = (size_t)brow * H_ + col;
            const float ig = sigmoid_(acc[0][v]);
            const float fg = sigmoid_(acc[1][v]);
            const float og = sigmoid_(acc[2][v]);
            const float gg = tanh_(acc[3][v]);
            const float cn = fg * cbuf[idx] + ig * gg;   // exclusive per wave
            cbuf[idx] = cn;
            const float hv = og * tanh_(cn);
            out[(size_t)t * B_ * H_ + idx] = hv;
            hout[idx] = (_Float16)hv;
            if (t == T_ - 1) {
                out[(size_t)T_ * B_ * H_ + idx]           = hv;  // final h
                out[(size_t)T_ * B_ * H_ + B_ * H_ + idx] = cn;  // final c
            }
        }

        // device-wide barrier (32 blocks are trivially co-resident)
        __threadfence();
        __syncthreads();
        if (threadIdx.x == 0) {
            __hip_atomic_fetch_add(cnt, 1u, __ATOMIC_RELEASE,
                                   __HIP_MEMORY_SCOPE_AGENT);
            const unsigned target = (unsigned)gridDim.x * (unsigned)(t + 1);
            while (__hip_atomic_load(cnt, __ATOMIC_ACQUIRE,
                                     __HIP_MEMORY_SCOPE_AGENT) < target) {
                __builtin_amdgcn_s_sleep(1);
            }
        }
        __syncthreads();
        __threadfence();
    }
}

// ----------------------------------------------------------------
extern "C" void kernel_launch(void* const* d_in, const int* in_sizes, int n_in,
                              void* d_out, int out_size, void* d_ws, size_t ws_size,
                              hipStream_t stream) {
    const float* x  = (const float*)d_in[0];  // batch_x        [T,B,I]
    const float* h0 = (const float*)d_in[1];  // initial_state  [B,H]
    const float* c0 = (const float*)d_in[2];  // initial_state_c[B,H]
    const float* Wi = (const float*)d_in[3];  // W_input        [T,I,4H]
    const float* bi = (const float*)d_in[4];  // B_input        [T,4H]
    const float* Wh = (const float*)d_in[5];  // W_hidden       [T,H,4H]
    const float* bh = (const float*)d_in[6];  // B_hidden       [T,4H]
    float* out = (float*)d_out;

    char*      ws   = (char*)d_ws;
    float*     Gp   = (float*)ws;                       // 64 MB packed pre-gates
    v16h*      Whp  = (v16h*)(ws + WS_WHP_OFF);         // 64 MB packed f16 Wh
    _Float16*  h16  = (_Float16*)(ws + WS_H16_OFF);     // f16 h, x2 buffers
    float*     cbuf = (float*)(ws + WS_C_OFF);          // f32 c state
    unsigned*  cnt  = (unsigned*)(ws + WS_CNT_OFF);     // grid-sync counter

    lstm_init_kernel<<<(B_ * H_ + 255) / 256, 256, 0, stream>>>(h0, c0, h16,
                                                                cbuf, cnt);
    // 65536 pack tiles / 4 waves per block
    lstm_pack_wh_kernel<<<16384, 128, 0, stream>>>(Wh, Whp);
    // 65536 pre-gate tiles / 4 waves per block
    lstm_pregate_kernel<<<16384, 128, 0, stream>>>(x, Wi, bi, bh, Gp);
    // persistent recurrent kernel: 32 blocks x 4 waves = 128 tile owners
    lstm_steps_kernel<<<32, 128, 0, stream>>>(Gp, Whp, h16, cbuf, out, cnt);
}
